// CRA_28278064677384
// MI455X (gfx1250) — matrix operand order repaired and checked by reading, not compiled, not run on hardware
//
#include <hip/hip_runtime.h>

// ---------------------------------------------------------------------------
// CDNA5 (gfx1250) implementation.
// Pipeline: chunk-mean gather -> bf16 convert + row norms -> WMMA bf16 GEMM
// with fused argmin epilogue -> codebook gather -> 2x WMMA GEMM MLP (bias/relu
// fused) -> 3x WMMA GEMM with fused RBF-sum epilogue -> final MMD combine.
// Matrix math: v_wmma_f32_16x16x32_bf16. Tile staging: double-buffered LDS
// fed by global_load_async_to_lds_b128 (ASYNCcnt), per CDNA5 ISA.
// ---------------------------------------------------------------------------

typedef __attribute__((ext_vector_type(16))) __bf16 v16bf;
typedef __attribute__((ext_vector_type(8)))  float  v8f;

// dims (match reference)
#define NCHUNK 8192   // B*NW = 16*512
#define DDIM   512
#define WCB    8192
#define MTEXT  4096
#define LWORD  8

__device__ __forceinline__ unsigned short f32_to_bf16(float f) {
    unsigned u = __float_as_uint(f);
    unsigned r = u + 0x7FFFu + ((u >> 16) & 1u);   // round-to-nearest-even
    return (unsigned short)(r >> 16);
}
__device__ __forceinline__ float bf16_to_f32(unsigned short h) {
    return __uint_as_float(((unsigned)h) << 16);
}

// --- gfx1250 async global->LDS copy (16B per lane), tracked by ASYNCcnt ----
// LDS operand of the instruction is the LDS byte address; per ISA 10.2 the
// low 32 bits of a generic shared pointer are exactly that address.
__device__ __forceinline__ void async_copy16(void* lds_dst, const void* gsrc) {
    unsigned lds_off = (unsigned)(unsigned long long)lds_dst;
    unsigned long long ga = (unsigned long long)gsrc;
    asm volatile("global_load_async_to_lds_b128 %0, %1, off"
                 :: "v"(lds_off), "v"(ga)
                 : "memory");
}
__device__ __forceinline__ void wait_async0() {
    asm volatile("s_wait_asynccnt 0x0" ::: "memory");
}

// ------------------------------ init ---------------------------------------
__global__ void init_k(unsigned long long* packed, double* acc) {
    int i = blockIdx.x * 256 + threadIdx.x;
    if (i < NCHUNK) packed[i] = ~0ull;
    if (i < 3) acc[i] = 0.0;
}

// ---------------------- chunk mean of char embeddings ----------------------
__global__ __launch_bounds__(128)
void chunk_mean_k(const int* __restrict__ toks, const float* __restrict__ cb,
                  unsigned short* __restrict__ ceb, float* __restrict__ cen) {
    __shared__ float red[128];
    const int i   = blockIdx.x;
    const int tid = threadIdx.x;
    const int c   = tid * 4;
    const int* t  = toks + (size_t)i * LWORD;
    float a0 = 0.f, a1 = 0.f, a2 = 0.f, a3 = 0.f;
    #pragma unroll
    for (int l = 0; l < LWORD; ++l) {
        const float* r = cb + (size_t)t[l] * DDIM + c;
        a0 += r[0]; a1 += r[1]; a2 += r[2]; a3 += r[3];
    }
    a0 *= 0.125f; a1 *= 0.125f; a2 *= 0.125f; a3 *= 0.125f;
    unsigned short* o = ceb + (size_t)i * DDIM + c;
    o[0] = f32_to_bf16(a0); o[1] = f32_to_bf16(a1);
    o[2] = f32_to_bf16(a2); o[3] = f32_to_bf16(a3);
    red[tid] = a0 * a0 + a1 * a1 + a2 * a2 + a3 * a3;
    __syncthreads();
    for (int s = 64; s > 0; s >>= 1) {
        if (tid < s) red[tid] += red[tid + s];
        __syncthreads();
    }
    if (tid == 0) cen[i] = red[0];
}

// -------------- fp32 rows -> bf16 + squared-norm per row -------------------
__global__ __launch_bounds__(128)
void rownorm_f32_k(const float* __restrict__ in, unsigned short* __restrict__ outb,
                   float* __restrict__ norms) {
    __shared__ float red[128];
    const int r = blockIdx.x, tid = threadIdx.x, c = tid * 4;
    const float* src = in + (size_t)r * DDIM + c;
    float a0 = src[0], a1 = src[1], a2 = src[2], a3 = src[3];
    unsigned short* o = outb + (size_t)r * DDIM + c;
    o[0] = f32_to_bf16(a0); o[1] = f32_to_bf16(a1);
    o[2] = f32_to_bf16(a2); o[3] = f32_to_bf16(a3);
    red[tid] = a0 * a0 + a1 * a1 + a2 * a2 + a3 * a3;
    __syncthreads();
    for (int s = 64; s > 0; s >>= 1) {
        if (tid < s) red[tid] += red[tid + s];
        __syncthreads();
    }
    if (tid == 0) norms[r] = red[0];
}

// -------------- bf16 rows -> squared-norm per row --------------------------
__global__ __launch_bounds__(128)
void rownorm_bf16_k(const unsigned short* __restrict__ in, float* __restrict__ norms) {
    __shared__ float red[128];
    const int r = blockIdx.x, tid = threadIdx.x, c = tid * 4;
    const unsigned short* src = in + (size_t)r * DDIM + c;
    float a0 = bf16_to_f32(src[0]), a1 = bf16_to_f32(src[1]);
    float a2 = bf16_to_f32(src[2]), a3 = bf16_to_f32(src[3]);
    red[tid] = a0 * a0 + a1 * a1 + a2 * a2 + a3 * a3;
    __syncthreads();
    for (int s = 64; s > 0; s >>= 1) {
        if (tid < s) red[tid] += red[tid + s];
        __syncthreads();
    }
    if (tid == 0) norms[r] = red[0];
}

// ---------------- W (DxD) -> W^T as bf16 (for A @ B^T form) ----------------
__global__ void transpose_cvt_k(const float* __restrict__ in, unsigned short* __restrict__ out) {
    int idx = blockIdx.x * 256 + threadIdx.x;      // 512*512 elems
    int k = idx >> 9, n = idx & 511;
    out[(size_t)n * DDIM + k] = f32_to_bf16(in[(size_t)k * DDIM + n]);
}

// ------------------- packed argmin key helpers -----------------------------
__device__ __forceinline__ unsigned long long pack_key(float d2, unsigned col) {
    d2 = fmaxf(d2, 0.0f);                          // non-neg => bits monotone
    return ((unsigned long long)__float_as_uint(d2) << 32) | (unsigned long long)col;
}
__device__ __forceinline__ unsigned long long halfwave_min_u64(unsigned long long key) {
    #pragma unroll
    for (int off = 1; off < 16; off <<= 1) {       // reduce across 16 col-lanes
        unsigned lo = __shfl_xor((unsigned)(key & 0xffffffffull), off, 16);
        unsigned hi = __shfl_xor((unsigned)(key >> 32), off, 16);
        unsigned long long o = ((unsigned long long)hi << 32) | (unsigned long long)lo;
        if (o < key) key = o;
    }
    return key;
}

// ---------------------------------------------------------------------------
// Tiled bf16 WMMA GEMM computing S = A(MxK) @ B(NxK)^T, f32 accumulate.
// Block tile 128x128, K-step 32; 8 waves (wave32): 4 along M x 2 along N,
// each wave owns 2x4 16x16 wmma accumulators. Double-buffered LDS tiles are
// filled by global_load_async_to_lds_b128; iteration t prefetches tile t+1
// while WMMAs consume tile t, then s_wait_asynccnt 0 + barrier.
// EPI=1: nearest-neighbour (d2 = rn + cn - 2S, packed atomicMin per row)
// EPI=2: out_bf16 = [relu](S + bias[col])
// EPI=3: atomicAdd(accum, sum(exp(-0.5*(rn + cn - 2S))))
// ---------------------------------------------------------------------------
template <int EPI>
__global__ __launch_bounds__(256)
void gemm_bf16_k(const unsigned short* __restrict__ A,
                 const unsigned short* __restrict__ B,
                 int M, int N, int K,
                 const float* __restrict__ rowNorm,
                 const float* __restrict__ colNorm,
                 const float* __restrict__ bias,
                 unsigned long long* __restrict__ packedMin,
                 unsigned short* __restrict__ outB,
                 double* __restrict__ accum,
                 int relu) {
    __shared__ unsigned short As[2][128 * 32];
    __shared__ unsigned short Bs[2][128 * 32];

    const int tid   = threadIdx.x;
    const int lane  = tid & 31;
    const int wave  = tid >> 5;
    const int waveM = wave & 3;       // 0..3
    const int waveN = wave >> 2;      // 0..1
    const int m0    = blockIdx.y * 128;
    const int n0    = blockIdx.x * 128;
    const int lr    = lane & 15;      // row/col index within 16
    const int half  = lane >> 4;      // lane half selects K sub-range

    v8f zero = {0.f, 0.f, 0.f, 0.f, 0.f, 0.f, 0.f, 0.f};
    v8f acc[2][4];
    #pragma unroll
    for (int i = 0; i < 2; ++i)
        #pragma unroll
        for (int j = 0; j < 4; ++j) acc[i][j] = zero;

    union Frag { uint4 q[2]; v16bf v; };

    // cooperative async tile fill: 512 chunks of 16B per matrix
    auto issue_tile = [&](int kb, int buf) {
        #pragma unroll
        for (int c = tid; c < 512; c += 256) {
            int row = c >> 2, kc = c & 3;
            async_copy16(&As[buf][row * 32 + kc * 8],
                         A + (size_t)(m0 + row) * K + kb + kc * 8);
            async_copy16(&Bs[buf][row * 32 + kc * 8],
                         B + (size_t)(n0 + row) * K + kb + kc * 8);
        }
    };

    const int nk = K >> 5;
    issue_tile(0, 0);
    wait_async0();
    __syncthreads();

    for (int t = 0; t < nk; ++t) {
        const int cur = t & 1;
        if (t + 1 < nk) issue_tile((t + 1) << 5, cur ^ 1);

        const unsigned short* Ab = As[cur];
        const unsigned short* Bb = Bs[cur];

        // A fragments: lane holds row (waveM*32 + mi*16 + lr),
        // K = half*8..+7 in VGPR0-3 and 16+half*8..+7 in VGPR4-7 (ISA 7.12.2)
        Frag af[2];
        #pragma unroll
        for (int mi = 0; mi < 2; ++mi) {
            int r = waveM * 32 + mi * 16 + lr;
            af[mi].q[0] = *reinterpret_cast<const uint4*>(&Ab[r * 32 + half * 8]);
            af[mi].q[1] = *reinterpret_cast<const uint4*>(&Ab[r * 32 + 16 + half * 8]);
        }
        // B fragments: lane holds col (waveN*64 + ni*16 + lr),
        // K = half*8..+7 in VGPR0-3 and 16+half*8..+7 in VGPR4-7 (same as A)
        Frag bf_[4];
        #pragma unroll
        for (int ni = 0; ni < 4; ++ni) {
            int n = waveN * 64 + ni * 16 + lr;
            bf_[ni].q[0] = *reinterpret_cast<const uint4*>(&Bb[n * 32 + half * 8]);
            bf_[ni].q[1] = *reinterpret_cast<const uint4*>(&Bb[n * 32 + 16 + half * 8]);
        }

        #pragma unroll
        for (int mi = 0; mi < 2; ++mi)
            #pragma unroll
            for (int ni = 0; ni < 4; ++ni)
                acc[mi][ni] = __builtin_amdgcn_wmma_f32_16x16x32_bf16(
                    false, af[mi].v, false, bf_[ni].v,
                    (short)0, acc[mi][ni], false, false);

        wait_async0();      // next tile resident before anyone passes barrier
        __syncthreads();
    }

    // ------------------------------ epilogues ------------------------------
    if (EPI == 1) {  // fused nearest-neighbour argmin
        #pragma unroll
        for (int mi = 0; mi < 2; ++mi) {
            int rowBase = m0 + waveM * 32 + mi * 16 + 8 * half;
            float cn[4]; int col[4];
            #pragma unroll
            for (int ni = 0; ni < 4; ++ni) {
                col[ni] = n0 + waveN * 64 + ni * 16 + lr;
                cn[ni]  = colNorm[col[ni]];
            }
            #pragma unroll
            for (int e = 0; e < 8; ++e) {
                int row  = rowBase + e;
                float rn = rowNorm[row];
                unsigned long long key = ~0ull;
                #pragma unroll
                for (int ni = 0; ni < 4; ++ni) {
                    float d2 = rn + cn[ni] - 2.0f * acc[mi][ni][e];
                    unsigned long long k2 = pack_key(d2, (unsigned)col[ni]);
                    if (k2 < key) key = k2;
                }
                key = halfwave_min_u64(key);
                if (lr == 0) atomicMin(&packedMin[row], key);
            }
        }
    } else if (EPI == 2) {  // bias (+relu) -> bf16 store
        #pragma unroll
        for (int mi = 0; mi < 2; ++mi) {
            int rowBase = m0 + waveM * 32 + mi * 16 + 8 * half;
            #pragma unroll
            for (int ni = 0; ni < 4; ++ni) {
                int col  = n0 + waveN * 64 + ni * 16 + lr;
                float bv = bias[col];
                #pragma unroll
                for (int e = 0; e < 8; ++e) {
                    float v = acc[mi][ni][e] + bv;
                    if (relu) v = fmaxf(v, 0.0f);
                    outB[(size_t)(rowBase + e) * N + col] = f32_to_bf16(v);
                }
            }
        }
    } else {  // EPI == 3: fused RBF sum (sigma = 1)
        float lsum = 0.0f;
        #pragma unroll
        for (int mi = 0; mi < 2; ++mi) {
            int rowBase = m0 + waveM * 32 + mi * 16 + 8 * half;
            #pragma unroll
            for (int ni = 0; ni < 4; ++ni) {
                int col  = n0 + waveN * 64 + ni * 16 + lr;
                float cn = colNorm[col];
                #pragma unroll
                for (int e = 0; e < 8; ++e) {
                    float d2 = rowNorm[rowBase + e] + cn - 2.0f * acc[mi][ni][e];
                    lsum += __expf(-0.5f * d2);
                }
            }
        }
        __syncthreads();                 // LDS tiles are dead; reuse as scratch
        float* red = reinterpret_cast<float*>(&As[0][0]);
        red[tid] = lsum;
        __syncthreads();
        for (int s = 128; s > 0; s >>= 1) {
            if (tid < s) red[tid] += red[tid + s];
            __syncthreads();
        }
        if (tid == 0) atomicAdd(accum, (double)red[0]);
    }
}

// ---------------- token select + codebook gather ---------------------------
__global__ __launch_bounds__(128)
void select_k(const unsigned long long* __restrict__ packed,
              const float* __restrict__ wc,
              float* __restrict__ outTok, float* __restrict__ outWe,
              unsigned short* __restrict__ web) {
    const int i   = blockIdx.x;
    const int tid = threadIdx.x;
    const unsigned tok = (unsigned)(packed[i] & 0xffffffffull);
    if (tid == 0) outTok[i] = (float)tok;
    const int c = tid * 4;
    const float* r = wc + (size_t)tok * DDIM + c;
    float a0 = r[0], a1 = r[1], a2 = r[2], a3 = r[3];
    float* o = outWe + (size_t)i * DDIM + c;
    o[0] = a0; o[1] = a1; o[2] = a2; o[3] = a3;
    unsigned short* ob = web + (size_t)i * DDIM + c;
    ob[0] = f32_to_bf16(a0); ob[1] = f32_to_bf16(a1);
    ob[2] = f32_to_bf16(a2); ob[3] = f32_to_bf16(a3);
}

// ---------------------------- final combine --------------------------------
__global__ void final_k(const double* __restrict__ acc, float* __restrict__ out) {
    double npp = (double)NCHUNK * (double)NCHUNK;
    double ntt = (double)MTEXT * (double)MTEXT;
    double npt = (double)NCHUNK * (double)MTEXT;
    out[0] = (float)(acc[0] / npp + acc[1] / ntt - 2.0 * acc[2] / npt);
}

// ---------------------------------------------------------------------------
extern "C" void kernel_launch(void* const* d_in, const int* in_sizes, int n_in,
                              void* d_out, int out_size, void* d_ws, size_t ws_size,
                              hipStream_t stream) {
    const int*   char_tokens = (const int*)  d_in[0];   // [16,4096]
    const float* char_cb     = (const float*)d_in[1];   // [1024,512]
    const float* word_cb     = (const float*)d_in[2];   // [8192,512]
    const float* W1          = (const float*)d_in[3];   // [512,512]
    const float* b1          = (const float*)d_in[4];   // [512]
    const float* W2          = (const float*)d_in[5];   // [512,512]
    const float* b2          = (const float*)d_in[6];   // [512]
    const float* text_emb    = (const float*)d_in[7];   // [4096,512]
    (void)in_sizes; (void)n_in; (void)out_size; (void)ws_size;

    // workspace carve-up (all offsets 256B-aligned)
    char* ws = (char*)d_ws;
    size_t off = 0;
    auto alloc = [&](size_t bytes) { char* p = ws + off; off = (off + bytes + 255) & ~(size_t)255; return p; };
    unsigned short* ce_b   = (unsigned short*)alloc((size_t)NCHUNK * DDIM * 2);
    unsigned short* wc_b   = (unsigned short*)alloc((size_t)WCB    * DDIM * 2);
    unsigned short* te_b   = (unsigned short*)alloc((size_t)MTEXT  * DDIM * 2);
    unsigned short* we_b   = (unsigned short*)alloc((size_t)NCHUNK * DDIM * 2);
    unsigned short* h_b    = (unsigned short*)alloc((size_t)NCHUNK * DDIM * 2);
    unsigned short* p_b    = (unsigned short*)alloc((size_t)NCHUNK * DDIM * 2);
    unsigned short* w1t_b  = (unsigned short*)alloc((size_t)DDIM * DDIM * 2);
    unsigned short* w2t_b  = (unsigned short*)alloc((size_t)DDIM * DDIM * 2);
    float* ce_n            = (float*)alloc(NCHUNK * 4);
    float* wc_n            = (float*)alloc(WCB    * 4);
    float* te_n            = (float*)alloc(MTEXT  * 4);
    float* p_n             = (float*)alloc(NCHUNK * 4);
    unsigned long long* pk = (unsigned long long*)alloc(NCHUNK * 8);
    double* accs           = (double*)alloc(3 * 8);

    float* out_tok = (float*)d_out;                       // 8192 tokens (as float)
    float* out_we  = out_tok + NCHUNK;                    // 8192*512 fp32
    float* out_mmd = out_we + (size_t)NCHUNK * DDIM;      // scalar

    // 1) init argmin keys + accumulators
    init_k<<<dim3((NCHUNK + 255) / 256), dim3(256), 0, stream>>>(pk, accs);
    // 2) chunk means -> ce (bf16) + norms
    chunk_mean_k<<<dim3(NCHUNK), dim3(128), 0, stream>>>(char_tokens, char_cb, ce_b, ce_n);
    // 3) convert codebooks / text to bf16 + norms
    rownorm_f32_k<<<dim3(WCB),   dim3(128), 0, stream>>>(word_cb,  wc_b, wc_n);
    rownorm_f32_k<<<dim3(MTEXT), dim3(128), 0, stream>>>(text_emb, te_b, te_n);
    transpose_cvt_k<<<dim3(DDIM * DDIM / 256), dim3(256), 0, stream>>>(W1, w1t_b);
    transpose_cvt_k<<<dim3(DDIM * DDIM / 256), dim3(256), 0, stream>>>(W2, w2t_b);
    // 4) nearest-codebook: ce @ wc^T with fused argmin (8192x8192x512)
    gemm_bf16_k<1><<<dim3(WCB / 128, NCHUNK / 128), dim3(256), 0, stream>>>(
        ce_b, wc_b, NCHUNK, WCB, DDIM, ce_n, wc_n, nullptr, pk, nullptr, nullptr, 0);
    // 5) gather selected rows -> tokens (d_out), we fp32 (d_out), we bf16 (ws)
    select_k<<<dim3(NCHUNK), dim3(128), 0, stream>>>(pk, word_cb, out_tok, out_we, we_b);
    // 6) MLP layer 1: relu(we @ W1 + b1) -> h (bf16)
    gemm_bf16_k<2><<<dim3(DDIM / 128, NCHUNK / 128), dim3(256), 0, stream>>>(
        we_b, w1t_b, NCHUNK, DDIM, DDIM, nullptr, nullptr, b1, nullptr, h_b, nullptr, 1);
    // 7) MLP layer 2: h @ W2 + b2 -> projected (bf16)
    gemm_bf16_k<2><<<dim3(DDIM / 128, NCHUNK / 128), dim3(256), 0, stream>>>(
        h_b, w2t_b, NCHUNK, DDIM, DDIM, nullptr, nullptr, b2, nullptr, p_b, nullptr, 0);
    // 8) projected norms
    rownorm_bf16_k<<<dim3(NCHUNK), dim3(128), 0, stream>>>(p_b, p_n);
    // 9) MMD RBF sums (fused into GEMM epilogue)
    gemm_bf16_k<3><<<dim3(NCHUNK / 128, NCHUNK / 128), dim3(256), 0, stream>>>(
        p_b, p_b, NCHUNK, NCHUNK, DDIM, p_n, p_n, nullptr, nullptr, nullptr, &accs[0], 0);
    gemm_bf16_k<3><<<dim3(MTEXT / 128, MTEXT / 128), dim3(256), 0, stream>>>(
        te_b, te_b, MTEXT, MTEXT, DDIM, te_n, te_n, nullptr, nullptr, nullptr, &accs[1], 0);
    gemm_bf16_k<3><<<dim3(MTEXT / 128, NCHUNK / 128), dim3(256), 0, stream>>>(
        p_b, te_b, NCHUNK, MTEXT, DDIM, p_n, te_n, nullptr, nullptr, nullptr, &accs[2], 0);
    // 10) combine
    final_k<<<dim3(1), dim3(1), 0, stream>>>(accs, out_mmd);
}
